// AttentionBlock_48163763257352
// MI455X (gfx1250) — matrix operand order repaired and checked by reading, not compiled, not run on hardware
//
#include <hip/hip_runtime.h>
#include <hip/hip_bf16.h>
#include <stdint.h>

// ---------------------------------------------------------------------------
// AttentionBlock for MI455X (gfx1250, wave32, v_wmma_f32_16x16x32_bf16)
// B=4, C=256, N=4096. Flash-attention streaming (no 256MB score tensor).
// K/V tiles via async global->LDS (ASYNCcnt), software-pipelined per wave.
// q,k interleaved in one [b][n][512] buffer -> packed b128 GEMM epilogue.
// ---------------------------------------------------------------------------

#define BATCH 4
#define CDIM  256
#define NTOK  4096
#define NGRP  8
#define QKW   (2 * CDIM)          // 512: row width of the fused q|k buffer

typedef __bf16 bf16_t;
typedef __attribute__((ext_vector_type(16))) __bf16 bf16x16;
typedef __attribute__((ext_vector_type(8)))  __bf16 bf16x8;
typedef __attribute__((ext_vector_type(8)))  float  f32x8;

__device__ __forceinline__ f32x8 wmma_bf16(bf16x16 a, bf16x16 b, f32x8 c) {
  return __builtin_amdgcn_wmma_f32_16x16x32_bf16(false, a, false, b,
                                                 (short)0, c, false, false);
}

// Build a 16-element fragment from two 16B regions (two b128 loads).
__device__ __forceinline__ bf16x16 ldfrag(const __bf16* p0, const __bf16* p1) {
  bf16x8 lo = *(const bf16x8*)p0;
  bf16x8 hi = *(const bf16x8*)p1;
  return __builtin_shufflevector(lo, hi, 0, 1, 2, 3, 4, 5, 6, 7,
                                 8, 9, 10, 11, 12, 13, 14, 15);
}
// A 16x32 fragment (ISA 7.12.2): lane half h reads K = [8h..8h+7], [16+8h..+7]
__device__ __forceinline__ bf16x16 ldfrag_a(const __bf16* base, int hf) {
  return ldfrag(base + 8 * hf, base + 16 + 8 * hf);
}
// B 32x16 fragment: lane half h reads K = [8h..8h+7], [16+8h..+7]
__device__ __forceinline__ bf16x16 ldfrag_b(const __bf16* base, int hf) {
  return ldfrag(base + 8 * hf, base + 16 + 8 * hf);
}

// ---------------------------------------------------------------------------
// Kernel 0: one-time fp32 -> bf16 conversion of qkv_w and proj_w.
// ---------------------------------------------------------------------------
__global__ __launch_bounds__(256)
void wconv_kernel(const float* __restrict__ qkv_w,
                  const float* __restrict__ proj_w,
                  bf16_t* __restrict__ wq_bf,
                  bf16_t* __restrict__ wp_bf) {
  const int idx = (blockIdx.x * 256 + threadIdx.x) * 4;
  const int NQ = 3 * CDIM * CDIM;          // 196608
  if (idx < NQ) {
    float4 f = *(const float4*)(qkv_w + idx);
    wq_bf[idx + 0] = (__bf16)f.x;
    wq_bf[idx + 1] = (__bf16)f.y;
    wq_bf[idx + 2] = (__bf16)f.z;
    wq_bf[idx + 3] = (__bf16)f.w;
  } else {
    const int j = idx - NQ;
    float4 f = *(const float4*)(proj_w + j);
    wp_bf[j + 0] = (__bf16)f.x;
    wp_bf[j + 1] = (__bf16)f.y;
    wp_bf[j + 2] = (__bf16)f.z;
    wp_bf[j + 3] = (__bf16)f.w;
  }
}

// ---------------------------------------------------------------------------
// Kernel 1: GroupNorm stats -> per (b,c) fused affine.
// ---------------------------------------------------------------------------
__global__ __launch_bounds__(256)
void gn_stats_kernel(const float* __restrict__ x,
                     const float* __restrict__ gn_w,
                     const float* __restrict__ gn_b,
                     float* __restrict__ cscale,
                     float* __restrict__ cshift) {
  __shared__ float s_sum[256];
  __shared__ float s_sq[256];
  __shared__ float s_mean, s_rstd;

  const int b = blockIdx.x >> 3;
  const int g = blockIdx.x & 7;
  const int t = threadIdx.x;
  const int cpg = CDIM / NGRP;
  const int cnt = cpg * NTOK;

  const float* xp = x + ((size_t)b * CDIM + (size_t)g * cpg) * NTOK;
  float sum = 0.f, sq = 0.f;
  for (int i = t; i < cnt; i += 256) {
    float v = xp[i];
    sum += v;
    sq  += v * v;
  }
  s_sum[t] = sum;
  s_sq[t]  = sq;
  __syncthreads();
  for (int off = 128; off > 0; off >>= 1) {
    if (t < off) {
      s_sum[t] += s_sum[t + off];
      s_sq[t]  += s_sq[t + off];
    }
    __syncthreads();
  }
  if (t == 0) {
    float mean = s_sum[0] / (float)cnt;
    float var  = s_sq[0] / (float)cnt - mean * mean;
    s_mean = mean;
    s_rstd = rsqrtf(var + 1e-5f);
  }
  __syncthreads();
  if (t < cpg) {
    int c = g * cpg + t;
    float sc = s_rstd * gn_w[c];
    cscale[b * CDIM + c] = sc;
    cshift[b * CDIM + c] = gn_b[c] - s_mean * sc;
  }
}

// ---------------------------------------------------------------------------
// Kernel 2: QKV GEMM. Stage normalized x tile to LDS once, B-fragments in
// registers, 8 waves x 6 o-tiles. Epilogue is tile-uniform: q/k tiles emit
// one packed bf16x8 (b128) store per lane into qk[b][n][512]; v tiles
// (o0 >= 512) store [b][c][n] for the P.V operand layout.
// ---------------------------------------------------------------------------
__global__ __launch_bounds__(256)
void qkv_kernel(const float* __restrict__ x,
                const bf16_t* __restrict__ wq_bf,
                const float* __restrict__ qkv_b,
                const float* __restrict__ cscale,
                const float* __restrict__ cshift,
                bf16_t* __restrict__ qk_t,
                bf16_t* __restrict__ v_out) {
  __shared__ __align__(16) __bf16 h_lds[16 * CDIM];   // [n][c], 8 KB

  const int n0   = blockIdx.x * 16;
  const int b    = blockIdx.z;
  const int wave = threadIdx.x >> 5;
  const int lane = threadIdx.x & 31;
  const int l16  = lane & 15;
  const int hf   = lane >> 4;

  {
    const int c = threadIdx.x;
    const float sc = cscale[b * CDIM + c];
    const float sh = cshift[b * CDIM + c];
    const float* xr = x + ((size_t)b * CDIM + c) * NTOK + n0;
    float4 f0 = *(const float4*)(xr + 0);
    float4 f1 = *(const float4*)(xr + 4);
    float4 f2 = *(const float4*)(xr + 8);
    float4 f3 = *(const float4*)(xr + 12);
    h_lds[0  * CDIM + c] = (__bf16)(f0.x * sc + sh);
    h_lds[1  * CDIM + c] = (__bf16)(f0.y * sc + sh);
    h_lds[2  * CDIM + c] = (__bf16)(f0.z * sc + sh);
    h_lds[3  * CDIM + c] = (__bf16)(f0.w * sc + sh);
    h_lds[4  * CDIM + c] = (__bf16)(f1.x * sc + sh);
    h_lds[5  * CDIM + c] = (__bf16)(f1.y * sc + sh);
    h_lds[6  * CDIM + c] = (__bf16)(f1.z * sc + sh);
    h_lds[7  * CDIM + c] = (__bf16)(f1.w * sc + sh);
    h_lds[8  * CDIM + c] = (__bf16)(f2.x * sc + sh);
    h_lds[9  * CDIM + c] = (__bf16)(f2.y * sc + sh);
    h_lds[10 * CDIM + c] = (__bf16)(f2.z * sc + sh);
    h_lds[11 * CDIM + c] = (__bf16)(f2.w * sc + sh);
    h_lds[12 * CDIM + c] = (__bf16)(f3.x * sc + sh);
    h_lds[13 * CDIM + c] = (__bf16)(f3.y * sc + sh);
    h_lds[14 * CDIM + c] = (__bf16)(f3.z * sc + sh);
    h_lds[15 * CDIM + c] = (__bf16)(f3.w * sc + sh);
  }
  __syncthreads();

  bf16x16 Bf[8];
#pragma unroll
  for (int kk = 0; kk < 8; ++kk)
    Bf[kk] = ldfrag_b(&h_lds[l16 * CDIM + kk * 32], hf);

  const int n = n0 + l16;
#pragma unroll
  for (int j = 0; j < 6; ++j) {
    const int o0 = (j * 8 + wave) * 16;           // 48 o-tiles
    f32x8 acc = {};
#pragma unroll
    for (int kk = 0; kk < 8; ++kk) {
      const bf16_t* wp = wq_bf + (size_t)(o0 + l16) * CDIM + kk * 32;
      acc = wmma_bf16(ldfrag_a(wp, hf), Bf[kk], acc);
    }
    // tile-uniform epilogue: rows o0+8hf .. o0+8hf+7 are consecutive
    if (o0 < 2 * CDIM) {                          // q or k tile -> packed b128
      const float scl = (o0 < CDIM) ? 0.0625f : 1.0f;  // C^-0.5 folded into q
      const float* bp = qkv_b + o0 + 8 * hf;
      bf16x8 pk;
#pragma unroll
      for (int v = 0; v < 8; ++v)
        pk[v] = (__bf16)((acc[v] + bp[v]) * scl);
      *(bf16x8*)(qk_t + ((size_t)b * NTOK + n) * QKW + o0 + 8 * hf) = pk;
    } else {                                      // v tile -> [b][c][n]
#pragma unroll
      for (int v = 0; v < 8; ++v) {
        const int o = o0 + v + 8 * hf;
        v_out[((size_t)b * CDIM + (o - 2 * CDIM)) * NTOK + n] =
            (__bf16)(acc[v] + qkv_b[o]);
      }
    }
  }
}

// ---------------------------------------------------------------------------
// Kernel 3: flash attention, 2 waves / 16-query tile, 32-key chunks.
// wave0 owns K async copies, wave1 owns V async copies (private ASYNCcnt):
// V loads overlap score WMMAs; K(kc+1) loads overlap P.V WMMAs.
// ---------------------------------------------------------------------------
__global__ __launch_bounds__(64)
void attn_kernel(const bf16_t* __restrict__ qk_t,
                 const bf16_t* __restrict__ v_in,
                 bf16_t* __restrict__ h_att) {
  __shared__ __align__(16) __bf16 k_lds[32 * CDIM];   // [key][c]  16 KB
  __shared__ __align__(16) __bf16 v_lds[CDIM * 32];   // [c][key]  16 KB
  __shared__ __align__(16) __bf16 p_lds[2][16 * 32];  // per-wave P

  const int b    = blockIdx.y;
  const int n0   = blockIdx.x * 16;
  const int wave = threadIdx.x >> 5;
  const int lane = threadIdx.x & 31;
  const int l16  = lane & 15;
  const int hf   = lane >> 4;
  const int NC   = NTOK / 32;

  // Q A-fragments: q half (offset 0) of the fused qk rows
  bf16x16 aq[8];
  const bf16_t* qrow = qk_t + ((size_t)b * NTOK + n0 + l16) * QKW;
#pragma unroll
  for (int kk = 0; kk < 8; ++kk)
    aq[kk] = ldfrag_a(qrow + kk * 32, hf);

  f32x8 O[8];
#pragma unroll
  for (int i = 0; i < 8; ++i) O[i] = (f32x8){};
  float mrow[8], lrow[8];
#pragma unroll
  for (int v = 0; v < 8; ++v) { mrow[v] = -1e30f; lrow[v] = 0.f; }

  const uint32_t kbase = (uint32_t)(size_t)&k_lds[0];
  const uint32_t vbase = (uint32_t)(size_t)&v_lds[0];
  const bf16_t* kb_g = qk_t + (size_t)b * NTOK * QKW + CDIM;  // k half
  const bf16_t* vb_g = v_in + (size_t)b * CDIM * NTOK;

  // ---- prologue: K(0) loaded, V(0) in flight ----
  if (wave == 0) {
    const char* gk = (const char*)(kb_g);
#pragma unroll
    for (int i = 0; i < 32; ++i) {          // 512B per key row, stride 1024B
      uint32_t laddr = kbase + i * 512 + lane * 16;
      uint64_t gaddr = (uint64_t)(gk + i * 1024 + lane * 16);
      asm volatile("global_load_async_to_lds_b128 %0, %1, off"
                   :: "v"(laddr), "v"(gaddr) : "memory");
    }
    asm volatile("s_wait_asynccnt 0" ::: "memory");
  } else {
    const char* gv = (const char*)(vb_g);
#pragma unroll
    for (int i = 0; i < 32; ++i) {
      const int c  = i * 8 + (lane >> 2);
      const int ko = (lane & 3) * 8;
      uint32_t laddr = vbase + c * 64 + ko * 2;
      uint64_t gaddr = (uint64_t)(gv + ((size_t)c * NTOK + ko) * 2);
      asm volatile("global_load_async_to_lds_b128 %0, %1, off"
                   :: "v"(laddr), "v"(gaddr) : "memory");
    }
  }
  __syncthreads();

#pragma unroll 1
  for (int kc = 0; kc < NC; ++kc) {
    // ---- scores (V[kc] still in flight on wave1's counter) ----
    f32x8 s0 = {}, s1 = {};
#pragma unroll
    for (int kk = 0; kk < 8; ++kk) {
      bf16x16 bk0 = ldfrag_b(&k_lds[(size_t)l16 * CDIM + kk * 32], hf);
      bf16x16 bk1 = ldfrag_b(&k_lds[(size_t)(16 + l16) * CDIM + kk * 32], hf);
      s0 = wmma_bf16(aq[kk], bk0, s0);
      s1 = wmma_bf16(aq[kk], bk1, s1);
    }

    // ---- online softmax; row = vgpr + 8*hf, reduce over 16-lane half ----
    float alpha[8];
#pragma unroll
    for (int v = 0; v < 8; ++v) {
      float mx = fmaxf(s0[v], s1[v]);
#pragma unroll
      for (int off = 1; off < 16; off <<= 1)
        mx = fmaxf(mx, __shfl_xor(mx, off, 32));
      const float mnew = fmaxf(mrow[v], mx);
      alpha[v] = __expf(mrow[v] - mnew);
      const float p0 = __expf(s0[v] - mnew);
      const float p1 = __expf(s1[v] - mnew);
      float rs = p0 + p1;
#pragma unroll
      for (int off = 1; off < 16; off <<= 1)
        rs += __shfl_xor(rs, off, 32);
      lrow[v] = lrow[v] * alpha[v] + rs;
      mrow[v] = mnew;
      s0[v] = p0;
      s1[v] = p1;
    }
#pragma unroll
    for (int i = 0; i < 8; ++i)
#pragma unroll
      for (int v = 0; v < 8; ++v) O[i][v] *= alpha[v];

    // ---- P: C-layout -> LDS -> A-layout (private per wave) ----
#pragma unroll
    for (int v = 0; v < 8; ++v) {
      const int row = v + 8 * hf;
      p_lds[wave][row * 32 + l16]      = (__bf16)s0[v];
      p_lds[wave][row * 32 + 16 + l16] = (__bf16)s1[v];
    }
    bf16x16 ap = ldfrag_a(&p_lds[wave][l16 * 32], hf);

    if (wave == 1)   // V[kc] complete (wave1's private ASYNCcnt)
      asm volatile("s_wait_asynccnt 0" ::: "memory");
    __syncthreads();  // V visible to both waves; k_lds readers done

    if (wave == 0 && kc + 1 < NC) {   // K(kc+1) overlaps P.V stage
      const char* gk = (const char*)(kb_g + (size_t)(kc + 1) * 32 * QKW);
#pragma unroll
      for (int i = 0; i < 32; ++i) {
        uint32_t laddr = kbase + i * 512 + lane * 16;
        uint64_t gaddr = (uint64_t)(gk + i * 1024 + lane * 16);
        asm volatile("global_load_async_to_lds_b128 %0, %1, off"
                     :: "v"(laddr), "v"(gaddr) : "memory");
      }
    }

    // ---- O += P . V^T : 8 channel chunks per wave ----
#pragma unroll
    for (int cc = 0; cc < 8; ++cc) {
      const int c = (wave * 8 + cc) * 16 + l16;
      bf16x16 bv = ldfrag_b(&v_lds[c * 32], hf);
      O[cc] = wmma_bf16(ap, bv, O[cc]);
    }
    __syncthreads();  // v_lds readers done

    if (kc + 1 < NC) {
      if (wave == 1) {   // V(kc+1) overlaps next chunk's score WMMAs
        const char* gv = (const char*)(vb_g + (size_t)(kc + 1) * 32);
#pragma unroll
        for (int i = 0; i < 32; ++i) {
          const int c  = i * 8 + (lane >> 2);
          const int ko = (lane & 3) * 8;
          uint32_t laddr = vbase + c * 64 + ko * 2;
          uint64_t gaddr = (uint64_t)(gv + ((size_t)c * NTOK + ko) * 2);
          asm volatile("global_load_async_to_lds_b128 %0, %1, off"
                       :: "v"(laddr), "v"(gaddr) : "memory");
        }
      }
      if (wave == 0)
        asm volatile("s_wait_asynccnt 0" ::: "memory");
    }
    __syncthreads();
  }

  bf16_t* hb = h_att + ((size_t)b * NTOK + n0) * CDIM;
#pragma unroll
  for (int v = 0; v < 8; ++v) {
    const float inv = 1.0f / lrow[v];
    const int row = v + 8 * hf;
#pragma unroll
    for (int cc = 0; cc < 8; ++cc) {
      const int c = (wave * 8 + cc) * 16 + l16;
      hb[(size_t)row * CDIM + c] = (__bf16)(O[cc][v] * inv);
    }
  }
}

// ---------------------------------------------------------------------------
// Kernel 4: out = x + proj_w . h_att + proj_b. h-tile async-staged to LDS,
// B-fragments in registers, 8 waves x 2 o-tiles per block.
// ---------------------------------------------------------------------------
__global__ __launch_bounds__(256)
void proj_kernel(const float* __restrict__ x,
                 const bf16_t* __restrict__ wp_bf,
                 const float* __restrict__ proj_b,
                 const bf16_t* __restrict__ h_att,
                 float* __restrict__ out) {
  __shared__ __align__(16) __bf16 h_lds[16 * CDIM];   // 8 KB

  const int n0   = blockIdx.x * 16;
  const int b    = blockIdx.z;
  const int wave = threadIdx.x >> 5;
  const int lane = threadIdx.x & 31;
  const int l16  = lane & 15;
  const int hf   = lane >> 4;

  const bf16_t* hb = h_att + ((size_t)b * NTOK + n0) * CDIM;
  if (wave == 0) {
    const uint32_t lbase = (uint32_t)(size_t)&h_lds[0];
#pragma unroll
    for (int i = 0; i < 16; ++i) {
      uint32_t laddr = lbase + i * 512 + lane * 16;
      uint64_t gaddr = (uint64_t)((const char*)hb + i * 512 + lane * 16);
      asm volatile("global_load_async_to_lds_b128 %0, %1, off"
                   :: "v"(laddr), "v"(gaddr) : "memory");
    }
    asm volatile("s_wait_asynccnt 0" ::: "memory");
  }
  __syncthreads();

  bf16x16 Bf[8];
#pragma unroll
  for (int kk = 0; kk < 8; ++kk)
    Bf[kk] = ldfrag_b(&h_lds[l16 * CDIM + kk * 32], hf);

  const int n = n0 + l16;
#pragma unroll
  for (int j = 0; j < 2; ++j) {
    const int o0 = (j * 8 + wave) * 16;            // 16 o-tiles
    f32x8 acc = {};
#pragma unroll
    for (int kk = 0; kk < 8; ++kk) {
      const bf16_t* wp = wp_bf + (size_t)(o0 + l16) * CDIM + kk * 32;
      acc = wmma_bf16(ldfrag_a(wp, hf), Bf[kk], acc);
    }
#pragma unroll
    for (int v = 0; v < 8; ++v) {
      const int o = o0 + v + 8 * hf;
      const size_t idx = ((size_t)b * CDIM + o) * NTOK + n;
      out[idx] = x[idx] + acc[v] + proj_b[o];
    }
  }
}

// ---------------------------------------------------------------------------
extern "C" void kernel_launch(void* const* d_in, const int* in_sizes, int n_in,
                              void* d_out, int out_size, void* d_ws, size_t ws_size,
                              hipStream_t stream) {
  const float* x      = (const float*)d_in[0];
  const float* gn_w   = (const float*)d_in[1];
  const float* gn_b   = (const float*)d_in[2];
  const float* qkv_w  = (const float*)d_in[3];
  const float* qkv_b  = (const float*)d_in[4];
  const float* proj_w = (const float*)d_in[5];
  const float* proj_b = (const float*)d_in[6];
  float* out = (float*)d_out;

  char* ws = (char*)d_ws;
  const size_t szAff = (size_t)BATCH * CDIM * sizeof(float);          // 4 KB
  const size_t szWq  = (size_t)3 * CDIM * CDIM * sizeof(bf16_t);      // 384 KB
  const size_t szWp  = (size_t)CDIM * CDIM * sizeof(bf16_t);          // 128 KB
  const size_t szQK  = (size_t)BATCH * NTOK * QKW * sizeof(bf16_t);   // 16 MB
  const size_t szMat = (size_t)BATCH * NTOK * CDIM * sizeof(bf16_t);  // 8 MB

  float*  cscale = (float*)(ws);
  float*  cshift = (float*)(ws + szAff);
  bf16_t* wq_bf  = (bf16_t*)(ws + 2 * szAff);
  bf16_t* wp_bf  = (bf16_t*)(ws + 2 * szAff + szWq);
  char*   base   = ws + 2 * szAff + szWq + szWp;
  bf16_t* qk_t   = (bf16_t*)(base);
  bf16_t* v_buf  = (bf16_t*)(base + szQK);
  bf16_t* h_att  = (bf16_t*)(base + szQK + szMat);

  wconv_kernel<<<dim3(256), 256, 0, stream>>>(qkv_w, proj_w, wq_bf, wp_bf);

  gn_stats_kernel<<<dim3(BATCH * NGRP), 256, 0, stream>>>(
      x, gn_w, gn_b, cscale, cshift);

  qkv_kernel<<<dim3(NTOK / 16, 1, BATCH), 256, 0, stream>>>(
      x, wq_bf, qkv_b, cscale, cshift, qk_t, v_buf);

  attn_kernel<<<dim3(NTOK / 16, BATCH), 64, 0, stream>>>(
      qk_t, v_buf, h_att);

  proj_kernel<<<dim3(NTOK / 16, 1, BATCH), 256, 0, stream>>>(
      x, wp_bf, proj_b, h_att, out);
}
